// Model_46901042872493
// MI455X (gfx1250) — compile-verified, matrix operand order checked by reading, not run on hardware
//
#include <hip/hip_runtime.h>
#include <hip/hip_bf16.h>
#include <string.h>

// ---------------- problem constants ----------------
#define Bn   128
#define Cn   512
#define Hn   16
#define Wn   64
#define An   256
#define HSn  512
#define INn  512
#define NCn  100
#define HWn  (Hn * Wn)            // 1024
#define BHW  (Bn * HWn)           // 131072

typedef __attribute__((ext_vector_type(16))) __bf16  v16bf;
typedef __attribute__((ext_vector_type(8)))  float   v8f;

union AFrag { v16bf v; unsigned int u[8]; };

// CDNA5 async global->LDS copies (ASYNCcnt), guarded so compile never breaks.
#if __has_builtin(__builtin_amdgcn_global_load_async_to_lds_b128) && \
    __has_builtin(__builtin_amdgcn_s_wait_asynccnt)
#define HAVE_ASYNC_LDS 1
#else
#define HAVE_ASYNC_LDS 0
#endif

#if HAVE_ASYNC_LDS
// Builtin params are pointers to a GCC generic vector: int __vector_size__(16),
// qualified AS1 (global / __device__) and AS3 (LDS / __shared__).
typedef int v4i_t __attribute__((vector_size(16)));
typedef __attribute__((address_space(1))) v4i_t g_v4i;
typedef __attribute__((address_space(3))) v4i_t l_v4i;
static __device__ __forceinline__ g_v4i* to_glb(const void* p) {
    return (g_v4i*)(unsigned long long)(uintptr_t)p;   // generic->AS1 via int
}
static __device__ __forceinline__ l_v4i* to_lds(const void* p) {
    return (l_v4i*)(unsigned int)(uintptr_t)p;         // generic->AS3 via int
}
#endif

static __device__ __forceinline__ unsigned short f2bf(float f) {
    unsigned u = __float_as_uint(f);
    u = (u + 0x7FFFu + ((u >> 16) & 1u)) >> 16;      // round-to-nearest-even
    return (unsigned short)u;
}

// K index inside a 16x32 bf16 A/B fragment for VGPR v, lane-half h (doc layout)
static __device__ __forceinline__ int kmap(int v, int h) {
    return (v < 4) ? (h * 8 + 2 * v) : (16 + h * 8 + 2 * (v - 4));
}

// ---------------- 1) features [B,C,H,W] f32 -> fT [B*H*W, C] bf16 -----------
__global__ void k_transpose_feat(const float* __restrict__ f,
                                 unsigned short* __restrict__ fT) {
    __shared__ float t[32][33];
    int w0 = blockIdx.x * 32, c0 = blockIdx.y * 32;
    int bh = blockIdx.z;                 // b*16 + h
    int b = bh >> 4, h = bh & 15;
    int tx = threadIdx.x, ty = threadIdx.y;   // (32, 8)
#pragma unroll
    for (int i = 0; i < 4; ++i) {
        int c = c0 + ty + i * 8;
        t[ty + i * 8][tx] = __builtin_nontemporal_load(
            &f[(((size_t)b * Cn + c) * Hn + h) * Wn + w0 + tx]);
    }
    __syncthreads();
#pragma unroll
    for (int i = 0; i < 4; ++i) {
        int wl  = ty + i * 8;
        int row = b * HWn + h * Wn + w0 + wl;
        fT[(size_t)row * Cn + c0 + tx] = f2bf(t[tx][wl]);
    }
}

// ---------------- 2) generic f32 -> bf16 convert ----------------------------
__global__ void k_f32_to_bf16(const float* __restrict__ s,
                              unsigned short* __restrict__ d, int n) {
    int i = blockIdx.x * 256 + threadIdx.x;
    if (i < n) d[i] = f2bf(s[i]);
}

// ---------------- 3) pack [x | h] into bf16 [128, 1024] ---------------------
__global__ void k_pack_xh(const float* __restrict__ x,
                          const float* __restrict__ hp,
                          unsigned short* __restrict__ xh) {
    int i = blockIdx.x * 256 + threadIdx.x;      // 131072 total
    int b = i >> 10, col = i & 1023;
    float v = (col < HSn) ? x[b * HSn + col] : hp[b * HSn + (col - HSn)];
    xh[i] = f2bf(v);
}

// ---------------- 4) WMMA bf16 GEMM: D[M,N] = A[M,K] @ B[K,N] (+epilogue) ---
// 256 threads (8 waves, 2x4), tile 128x128, K step 32, double-buffered LDS,
// async global->LDS for the A tile when the gfx1250 builtins are available.
// mode 1: D += ex[(row&1023)&63][n] + ey[(row&1023)>>6][n]

static __device__ __forceinline__ void stage_tile(
    const unsigned short* __restrict__ Ag,
    const unsigned short* __restrict__ Bg,
    unsigned short (*As)[40], unsigned short (*Bt)[40],
    int tid, int mBlk, int nBlk, int k0, int K, int N) {
#pragma unroll
    for (int it = 0; it < 2; ++it) {
        int idx = tid + it * 256;                          // 0..511
        int rA = idx >> 2, sA = idx & 3;
        const unsigned short* gA = Ag + (size_t)(mBlk + rA) * K + k0 + sA * 8;
#if HAVE_ASYNC_LDS
        __builtin_amdgcn_global_load_async_to_lds_b128(
            to_glb(gA), to_lds(&As[rA][sA * 8]), 0, 0);
#else
        *(uint4*)&As[rA][sA * 8] = *(const uint4*)gA;
#endif
        int kB = idx >> 4, sB = idx & 15;                  // B: transpose in regs
        union { uint4 q; unsigned short s[8]; } ub;
        ub.q = *(const uint4*)(Bg + (size_t)(k0 + kB) * N + nBlk + sB * 8);
#pragma unroll
        for (int e = 0; e < 8; ++e) Bt[sB * 8 + e][kB] = ub.s[e];
    }
}

__global__ __launch_bounds__(256)
void k_wmma_gemm(const unsigned short* __restrict__ Ag,
                 const unsigned short* __restrict__ Bg,
                 float* __restrict__ D,
                 int M, int N, int K,
                 const float* __restrict__ ex, const float* __restrict__ ey,
                 int mode) {
    __shared__ __attribute__((aligned(16))) unsigned short As[2][128][40];
    __shared__ __attribute__((aligned(16))) unsigned short Bt[2][128][40];

    const int tid    = threadIdx.x;
    const int wave   = tid >> 5;
    const int lane   = tid & 31;
    const int r      = lane & 15;
    const int h      = lane >> 4;
    const int wave_m = wave >> 2;            // 0..1 -> 64-row slabs
    const int wave_n = wave & 3;             // 0..3 -> 32-col slabs
    const int nBlk   = blockIdx.x * 128;
    const int mBlk   = blockIdx.y * 128;

    v8f acc[4][2];
#pragma unroll
    for (int ms = 0; ms < 4; ++ms)
#pragma unroll
        for (int ns = 0; ns < 2; ++ns)
            acc[ms][ns] = (v8f){0.f, 0.f, 0.f, 0.f, 0.f, 0.f, 0.f, 0.f};

    const int NT = K >> 5;                   // K / 32 tiles
    stage_tile(Ag, Bg, As[0], Bt[0], tid, mBlk, nBlk, 0, K, N);

    for (int t = 0; t < NT; ++t) {
        const int cur = t & 1;
#if HAVE_ASYNC_LDS
        __builtin_amdgcn_s_wait_asynccnt(0); // async fills of As[cur] done
#endif
        __syncthreads();                     // all stores to buf[cur] visible

        if (t + 1 < NT)                      // prefetch next tile into other buf
            stage_tile(Ag, Bg, As[1 - cur], Bt[1 - cur],
                       tid, mBlk, nBlk, (t + 1) << 5, K, N);

        // ---- fragments from buf[cur] ----
        AFrag af[4], bf[2];
#pragma unroll
        for (int ms = 0; ms < 4; ++ms) {
            int rowl = wave_m * 64 + ms * 16 + r;
#pragma unroll
            for (int v = 0; v < 8; ++v)
                af[ms].u[v] = *(const unsigned int*)&As[cur][rowl][kmap(v, h)];
        }
#pragma unroll
        for (int ns = 0; ns < 2; ++ns) {
            int coll = wave_n * 32 + ns * 16 + r;
#pragma unroll
            for (int v = 0; v < 8; ++v)
                bf[ns].u[v] = *(const unsigned int*)&Bt[cur][coll][kmap(v, h)];
        }

        // ---- 8 WMMAs per wave per K step ----
#pragma unroll
        for (int ms = 0; ms < 4; ++ms)
#pragma unroll
            for (int ns = 0; ns < 2; ++ns)
                acc[ms][ns] = __builtin_amdgcn_wmma_f32_16x16x32_bf16(
                    false, af[ms].v, false, bf[ns].v,
                    (short)0, acc[ms][ns], false, false);
    }

    // ---- epilogue + store (C/D layout: VGPR j -> row h*8+j, col r) ----
#pragma unroll
    for (int ms = 0; ms < 4; ++ms)
#pragma unroll
        for (int ns = 0; ns < 2; ++ns) {
            int col = nBlk + wave_n * 32 + ns * 16 + r;
#pragma unroll
            for (int j = 0; j < 8; ++j) {
                int row = mBlk + wave_m * 64 + ms * 16 + h * 8 + j;
                float v = acc[ms][ns][j];
                if (mode == 1) {
                    int p = row & (HWn - 1);
                    v += ex[(p & (Wn - 1)) * N + col] + ey[(p >> 6) * N + col];
                }
                D[(size_t)row * N + col] = v;
            }
        }
}

// ---------------- 5) ws = h @ Ws  ([128,512]@[512,256]) ---------------------
__global__ void k_proj_ws(const float* __restrict__ hst,
                          const float* __restrict__ Ws,
                          float* __restrict__ wsout) {
    int b = blockIdx.x, a = threadIdx.x;     // 256 threads
    float acc = 0.f;
    for (int k = 0; k < HSn; ++k) acc += hst[b * HSn + k] * Ws[k * An + a];
    wsout[b * An + a] = acc;
}

// ---------------- 6) scores[b,p] = tanh(wfxy + ws) . Wm ---------------------
__global__ void k_attn_scores(const float* __restrict__ wfxy,
                              const float* __restrict__ wsb,
                              const float* __restrict__ Wm,
                              float* __restrict__ scores) {
    int wave = threadIdx.x >> 5, lane = threadIdx.x & 31;
    int row = blockIdx.x * 8 + wave;         // 0..131071
    int b = row >> 10;
    float acc = 0.f;
#pragma unroll
    for (int i = 0; i < 8; ++i) {
        int a = lane + i * 32;
        acc += tanhf(wfxy[(size_t)row * An + a] + wsb[b * An + a]) * Wm[a];
    }
#pragma unroll
    for (int off = 16; off > 0; off >>= 1) acc += __shfl_xor(acc, off, 32);
    if (lane == 0) scores[row] = acc;
}

// ---------------- 7) softmax over H*W=1024 per batch ------------------------
__global__ void k_softmax_hw(const float* __restrict__ scores,
                             float* __restrict__ mask) {
    int b = blockIdx.x, tid = threadIdx.x;   // 256 threads
    __shared__ float red[256];
    float v[4], m = -1e30f;
#pragma unroll
    for (int i = 0; i < 4; ++i) {
        v[i] = scores[b * HWn + tid + i * 256];
        m = fmaxf(m, v[i]);
    }
    red[tid] = m; __syncthreads();
    for (int s = 128; s > 0; s >>= 1) {
        if (tid < s) red[tid] = fmaxf(red[tid], red[tid + s]);
        __syncthreads();
    }
    m = red[0]; __syncthreads();
    float sum = 0.f;
#pragma unroll
    for (int i = 0; i < 4; ++i) { v[i] = expf(v[i] - m); sum += v[i]; }
    red[tid] = sum; __syncthreads();
    for (int s = 128; s > 0; s >>= 1) {
        if (tid < s) red[tid] += red[tid + s];
        __syncthreads();
    }
    float inv = 1.f / red[0];
#pragma unroll
    for (int i = 0; i < 4; ++i) mask[b * HWn + tid + i * 256] = v[i] * inv;
}

// ---------------- 8) wfeat[b,c] = sum_p f[b,c,p] * mask[b,p] ----------------
__global__ void k_weighted_feat(const float* __restrict__ f,
                                const float* __restrict__ mask,
                                float* __restrict__ wfeat) {
    int wave = threadIdx.x >> 5, lane = threadIdx.x & 31;
    int idx = blockIdx.x * 8 + wave;         // b*512 + c, 65536 total
    int b = idx >> 9;
    const float* fp = f + (size_t)idx * HWn;
    const float* mp = mask + b * HWn;
    float acc = 0.f;
#pragma unroll 4
    for (int i = 0; i < 32; ++i) {
        int p = lane + i * 32;
        acc += __builtin_nontemporal_load(&fp[p]) * mp[p];   // stream 256MB, spare L2
    }
#pragma unroll
    for (int off = 16; off > 0; off >>= 1) acc += __shfl_xor(acc, off, 32);
    if (lane == 0) wfeat[idx] = acc;
}

// ---------------- 9) lstm_in = onehot(char)@Wchar + wfeat@Wfeat1 ------------
__global__ void k_lstm_in(const int* __restrict__ prev_char,
                          const float* __restrict__ Wchar,
                          const float* __restrict__ wfeat,
                          const float* __restrict__ Wfeat1,
                          float* __restrict__ out) {
    int b = blockIdx.x, tid = threadIdx.x;
    int pc = prev_char[b];
    for (int j = tid; j < INn; j += 256) {
        float acc = Wchar[pc * INn + j];
        for (int c = 0; c < Cn; ++c) acc += wfeat[b * Cn + c] * Wfeat1[c * INn + j];
        out[b * INn + j] = acc;
    }
}

// ---------------- 10) LSTM pointwise ----------------------------------------
__global__ void k_lstm_point(const float* __restrict__ g,
                             const float* __restrict__ bih,
                             const float* __restrict__ bhh,
                             const float* __restrict__ cprev,
                             float* __restrict__ hout,
                             float* __restrict__ cout) {
    int i = blockIdx.x * 256 + threadIdx.x;  // 65536
    int b = i >> 9, u = i & 511;
    const float* gb = g + (size_t)b * 4 * HSn;
    float gi = gb[u]            + bih[u]            + bhh[u];
    float gf = gb[HSn + u]      + bih[HSn + u]      + bhh[HSn + u];
    float gg = gb[2 * HSn + u]  + bih[2 * HSn + u]  + bhh[2 * HSn + u];
    float go = gb[3 * HSn + u]  + bih[3 * HSn + u]  + bhh[3 * HSn + u];
    float si = 1.f / (1.f + expf(-gi));
    float sf = 1.f / (1.f + expf(-gf));
    float so = 1.f / (1.f + expf(-go));
    float c2 = sf * cprev[i] + si * tanhf(gg);
    hout[i] = so * tanhf(c2);
    cout[i] = c2;
}

// ---------------- 11) logits + softmax probs --------------------------------
__global__ void k_logits_probs(const float* __restrict__ wfeat2,
                               const float* __restrict__ Wfeat2,
                               const float* __restrict__ h1,
                               const float* __restrict__ Wlstm,
                               float* __restrict__ probs) {
    int b = blockIdx.x, tid = threadIdx.x;   // 128 threads
    __shared__ float red[128];
    float l = 0.f;
    if (tid < NCn) {
        for (int c = 0; c < Cn; ++c)  l += wfeat2[b * Cn + c] * Wfeat2[c * NCn + tid];
        for (int k = 0; k < HSn; ++k) l += h1[b * HSn + k]    * Wlstm[k * NCn + tid];
    }
    red[tid] = (tid < NCn) ? l : -1e30f; __syncthreads();
    for (int s = 64; s > 0; s >>= 1) {
        if (tid < s) red[tid] = fmaxf(red[tid], red[tid + s]);
        __syncthreads();
    }
    float m = red[0]; __syncthreads();
    float e = (tid < NCn) ? expf(l - m) : 0.f;
    red[tid] = e; __syncthreads();
    for (int s = 64; s > 0; s >>= 1) {
        if (tid < s) red[tid] += red[tid + s];
        __syncthreads();
    }
    float inv = 1.f / red[0];
    if (tid < NCn) probs[b * NCn + tid] = e * inv;
}

// ============================================================================
extern "C" void kernel_launch(void* const* d_in, const int* in_sizes, int n_in,
                              void* d_out, int out_size, void* d_ws, size_t ws_size,
                              hipStream_t stream) {
    const float* features = (const float*)d_in[0];
    const float* prev_h   = (const float*)d_in[1];   // [2,128,512]
    const float* prev_c   = (const float*)d_in[2];
    const float* Wf       = (const float*)d_in[3];
    const float* Ws       = (const float*)d_in[4];
    const float* Wx       = (const float*)d_in[5];   // [64,256]
    const float* Wy       = (const float*)d_in[6];   // [16,256]
    const float* Wm       = (const float*)d_in[7];   // [256]
    const float* Wchar    = (const float*)d_in[8];
    const float* Wfeat1   = (const float*)d_in[9];
    const float* Wfeat2   = (const float*)d_in[10];
    const float* Wlstm    = (const float*)d_in[11];
    const float* Wih0     = (const float*)d_in[12];
    const float* Whh0     = (const float*)d_in[13];
    const float* bih0     = (const float*)d_in[14];
    const float* bhh0     = (const float*)d_in[15];
    const float* Wih1     = (const float*)d_in[16];
    const float* Whh1     = (const float*)d_in[17];
    const float* bih1     = (const float*)d_in[18];
    const float* bhh1     = (const float*)d_in[19];
    const int*   prev_ch  = (const int*)d_in[20];

    // -------- workspace bump allocator --------
    char* wp = (char*)d_ws;
    auto alloc = [&](size_t bytes) -> void* {
        void* p = wp; wp += (bytes + 255) & ~(size_t)255; return p;
    };
    unsigned short* fT     = (unsigned short*)alloc((size_t)BHW * Cn * 2);
    unsigned short* Wfb    = (unsigned short*)alloc((size_t)Cn * An * 2);
    float*          wfxy   = (float*)alloc((size_t)BHW * An * 4);
    unsigned short* Wcat0  = (unsigned short*)alloc((size_t)1024 * 2048 * 2);
    unsigned short* Wcat1  = (unsigned short*)alloc((size_t)1024 * 2048 * 2);
    unsigned short* xh     = (unsigned short*)alloc((size_t)Bn * 1024 * 2);
    float*          gates  = (float*)alloc((size_t)Bn * 2048 * 4);
    float*          wsb    = (float*)alloc((size_t)Bn * An * 4);
    float*          scores = (float*)alloc((size_t)BHW * 4);
    float*          mask1  = (float*)alloc((size_t)BHW * 4);
    float*          wft1   = (float*)alloc((size_t)Bn * Cn * 4);
    float*          lstmin = (float*)alloc((size_t)Bn * INn * 4);
    float*          wft2   = (float*)alloc((size_t)Bn * Cn * 4);

    // -------- output layout: probs, cur_h, cur_c, mask --------
    float* out   = (float*)d_out;
    float* probs = out;                                  // 128*100
    float* cur_h = out + Bn * NCn;                       // 2*128*512
    float* cur_c = cur_h + 2 * Bn * HSn;
    float* maskO = cur_c + 2 * Bn * HSn;                 // 128*1024
    float* h0 = cur_h;
    float* h1 = cur_h + Bn * HSn;
    float* c0 = cur_c;
    float* c1 = cur_c + Bn * HSn;
    const float* ph0 = prev_h;                 // layer 0
    const float* ph1 = prev_h + Bn * HSn;      // layer 1 ( == prev_h[-1] )
    const float* pc0 = prev_c;
    const float* pc1 = prev_c + Bn * HSn;

    // 1) feature transpose to bf16 + weight converts
    k_transpose_feat<<<dim3(Wn / 32, Cn / 32, Bn * Hn), dim3(32, 8), 0, stream>>>(features, fT);
    k_f32_to_bf16<<<(Cn * An + 255) / 256, 256, 0, stream>>>(Wf, Wfb, Cn * An);
    k_f32_to_bf16<<<(512 * 2048 + 255) / 256, 256, 0, stream>>>(Wih0, Wcat0, 512 * 2048);
    k_f32_to_bf16<<<(512 * 2048 + 255) / 256, 256, 0, stream>>>(Whh0, Wcat0 + 512 * 2048, 512 * 2048);
    k_f32_to_bf16<<<(512 * 2048 + 255) / 256, 256, 0, stream>>>(Wih1, Wcat1, 512 * 2048);
    k_f32_to_bf16<<<(512 * 2048 + 255) / 256, 256, 0, stream>>>(Whh1, Wcat1 + 512 * 2048, 512 * 2048);

    // 2) big attention GEMM (shared by both attention passes): wfxy = fT@Wf + wex + wey
    k_wmma_gemm<<<dim3(An / 128, BHW / 128), 256, 0, stream>>>(
        fT, Wfb, wfxy, BHW, An, Cn, Wx, Wy, 1);

    // 3) attention #1 with h = prev_h[1]
    k_proj_ws<<<Bn, An, 0, stream>>>(ph1, Ws, wsb);
    k_attn_scores<<<BHW / 8, 256, 0, stream>>>(wfxy, wsb, Wm, scores);
    k_softmax_hw<<<Bn, 256, 0, stream>>>(scores, mask1);
    k_weighted_feat<<<Bn * Cn / 8, 256, 0, stream>>>(features, mask1, wft1);

    // 4) lstm_in, LSTM layer 0
    k_lstm_in<<<Bn, 256, 0, stream>>>(prev_ch, Wchar, wft1, Wfeat1, lstmin);
    k_pack_xh<<<Bn * 1024 / 256, 256, 0, stream>>>(lstmin, ph0, xh);
    k_wmma_gemm<<<dim3(2048 / 128, 1), 256, 0, stream>>>(
        xh, Wcat0, gates, Bn, 2048, 1024, nullptr, nullptr, 0);
    k_lstm_point<<<Bn * HSn / 256, 256, 0, stream>>>(gates, bih0, bhh0, pc0, h0, c0);

    // 5) LSTM layer 1
    k_pack_xh<<<Bn * 1024 / 256, 256, 0, stream>>>(h0, ph1, xh);
    k_wmma_gemm<<<dim3(2048 / 128, 1), 256, 0, stream>>>(
        xh, Wcat1, gates, Bn, 2048, 1024, nullptr, nullptr, 0);
    k_lstm_point<<<Bn * HSn / 256, 256, 0, stream>>>(gates, bih1, bhh1, pc1, h1, c1);

    // 6) attention #2 with h1 (mask goes to output)
    k_proj_ws<<<Bn, An, 0, stream>>>(h1, Ws, wsb);
    k_attn_scores<<<BHW / 8, 256, 0, stream>>>(wfxy, wsb, Wm, scores);
    k_softmax_hw<<<Bn, 256, 0, stream>>>(scores, maskO);
    k_weighted_feat<<<Bn * Cn / 8, 256, 0, stream>>>(features, maskO, wft2);

    // 7) logits + softmax
    k_logits_probs<<<Bn, 128, 0, stream>>>(wft2, Wfeat2, h1, Wlstm, probs);
}
